// BronxLayer_45380624450026
// MI455X (gfx1250) — matrix operand order repaired and checked by reading, not compiled, not run on hardware
//
#include <hip/hip_runtime.h>
#include <hip/hip_bf16.h>

// ---------------------------------------------------------------------------
// Types for CDNA5 WMMA (wave32)
// ---------------------------------------------------------------------------
typedef __attribute__((ext_vector_type(16))) _Float16 v16h;
typedef __attribute__((ext_vector_type(8)))  _Float16 v8h;
typedef __attribute__((ext_vector_type(8)))  float    v8f;
typedef __attribute__((ext_vector_type(4)))  int      v4i;

#define NROWS 4096
#define INF   512
#define HIDC  512
#define DXC   256
#define HCPAD 576   // 512 hn + 3 stats + 61 zero pad (multiple of 64)

#define LDSROW 72   // f16 elems per LDS B row (64 data + 8 pad, bank-conflict-free)

#if __has_builtin(__builtin_amdgcn_global_load_async_to_lds_b128)
#define HAS_ASYNC_LDS 1
#else
#define HAS_ASYNC_LDS 0
#endif

// 16B global -> LDS copy; async DMA (ASYNCcnt) when available.
__device__ __forceinline__ void stage16(const _Float16* __restrict__ g,
                                        _Float16* __restrict__ l) {
#if HAS_ASYNC_LDS
  __builtin_amdgcn_global_load_async_to_lds_b128(
      (__attribute__((address_space(1))) v4i*)g,
      (__attribute__((address_space(3))) v4i*)l, 0, 0);
#else
  *(v8h*)l = *(const v8h*)g;
#endif
}

__device__ __forceinline__ void wait_stage() {
#if HAS_ASYNC_LDS
#if __has_builtin(__builtin_amdgcn_s_wait_asynccnt)
  __builtin_amdgcn_s_wait_asynccnt(0);
#endif
#endif
}

// ---------------------------------------------------------------------------
// WMMA A fragment (16-bit A 16x32 layout, wave32):
// lane m = lane&15; lanes<16 hold K 0..7 & 16..23, lanes>=16 hold 8..15 & 24..31
// ---------------------------------------------------------------------------
__device__ __forceinline__ v16h load_a_frag(const _Float16* __restrict__ rowp,
                                            int k0, int lane) {
  const int ksel = (lane >> 4) << 3;
  const _Float16* p = rowp + k0 + ksel;
  v8h lo = *(const v8h*)(p);
  v8h hi = *(const v8h*)(p + 16);
  v16h r;
#pragma unroll
  for (int i = 0; i < 8; ++i) { r[i] = lo[i]; r[i + 8] = hi[i]; }
  return r;
}

// B fragment from LDS: two 16B chunks per lane (K 0..7|8..15 and +16), two b128 loads
__device__ __forceinline__ v16h load_b_frag(const _Float16* __restrict__ p) {
  v8h lo = *(const v8h*)(p);
  v8h hi = *(const v8h*)(p + 16);
  v16h r;
#pragma unroll
  for (int i = 0; i < 8; ++i) { r[i] = lo[i]; r[i + 8] = hi[i]; }
  return r;
}

// ---------------------------------------------------------------------------
// WMMA GEMM: C[M,N] = A[M,K] * Bt[N,K]^T   (f16 in, f32 accumulate)
// Block = 128 threads = 4 wave32; block tile 128x128; wave tile 32x128.
// K stepped in blocks of 64: B tile (128 rows x 64 K) staged in LDS
// (double-buffered, async DMA, padded rows); 32 WMMAs per wave per K-block;
// B fragments software-pipelined one step ahead within the unrolled loop.
// Store modes:
//   0: f32   Cf[row*ldc + c*colScale + colOff]
//   1: f16   Ch[row*ldc + c*colScale + colOff]
//   2: f16 head-scatter   Ch[((c&3)*4096 + row)*128 + (c>>2)]         (K/Q)
//   3: f16 head-transpose Ch[((c&3)*128 + (c>>2))*4096 + row]         (V^T)
// ---------------------------------------------------------------------------
__global__ __launch_bounds__(128) void wmma_gemm(
    const _Float16* __restrict__ A,  int lda,
    const _Float16* __restrict__ Bt, int ldb,
    int Kdim, int mode,
    float* __restrict__ Cf, _Float16* __restrict__ Ch,
    int ldc, int colScale, int colOff)
{
  __shared__ __align__(64) _Float16 bsm[2][128 * LDSROW];

  const int tid   = threadIdx.x;
  const int lane  = tid & 31;
  const int wave  = tid >> 5;
  const int tileM = blockIdx.x * 128 + wave * 32;
  const int tileN = blockIdx.y * 128;
  const int mrow  = lane & 15;

  // staging role: thread t copies B row t (64 f16 = 8 x 16B chunks)
  const _Float16* gstage = Bt + (size_t)(tileN + tid) * ldb;
  _Float16* lstage0 = &bsm[0][tid * LDSROW];
  _Float16* lstage1 = &bsm[1][tid * LDSROW];

  const _Float16* aRow0 = A + (size_t)(tileM + mrow) * lda;
  const _Float16* aRow1 = aRow0 + (size_t)16 * lda;

  // prologue: stage B(kb=0) into buffer 0
#pragma unroll
  for (int c = 0; c < 8; ++c) stage16(gstage + c * 8, lstage0 + c * 8);

  v8f acc[16] = {};
  int cur = 0;

  for (int kb = 0; kb < Kdim; kb += 64) {
    wait_stage();
    __syncthreads();

    if (kb + 64 < Kdim) {
      // stage next K-block into the other buffer (overlaps with compute)
      const _Float16* gs = gstage + kb + 64;
      _Float16* ls = cur ? lstage0 : lstage1;
#pragma unroll
      for (int c = 0; c < 8; ++c) stage16(gs + c * 8, ls + c * 8);
      __builtin_prefetch(aRow0 + kb + 64, 0, 1);   // A for next block
    }

    // A fragments for this K-block (L2-resident; loads overlap ds reads)
    v16h a00 = load_a_frag(aRow0, kb,      lane);
    v16h a10 = load_a_frag(aRow1, kb,      lane);
    v16h a01 = load_a_frag(aRow0, kb + 32, lane);
    v16h a11 = load_a_frag(aRow1, kb + 32, lane);

    const _Float16* bb = &bsm[cur][0];
    const int bofs = (lane >> 4) << 3;   // K-half select within sub-step

    // software-pipelined B fragments: load j+1 while computing j
    const _Float16* bp0 = bb + mrow * LDSROW + bofs;
    v16h b0 = load_b_frag(bp0);
    v16h b1 = load_b_frag(bp0 + 32);
#pragma unroll
    for (int j = 0; j < 8; ++j) {
      v16h nb0 = b0, nb1 = b1;
      if (j < 7) {
        const _Float16* bpn = bb + ((j + 1) * 16 + mrow) * LDSROW + bofs;
        nb0 = load_b_frag(bpn);
        nb1 = load_b_frag(bpn + 32);
      }
      acc[j]     = __builtin_amdgcn_wmma_f32_16x16x32_f16(false, a00, false, b0, (short)0, acc[j],     false, false);
      acc[8 + j] = __builtin_amdgcn_wmma_f32_16x16x32_f16(false, a10, false, b0, (short)0, acc[8 + j], false, false);
      acc[j]     = __builtin_amdgcn_wmma_f32_16x16x32_f16(false, a01, false, b1, (short)0, acc[j],     false, false);
      acc[8 + j] = __builtin_amdgcn_wmma_f32_16x16x32_f16(false, a11, false, b1, (short)0, acc[8 + j], false, false);
      b0 = nb0; b1 = nb1;
    }
    cur ^= 1;
  }

  // C/D layout: VGPR r holds M = base + r + 8*(lane>=16), N = tileN + (lane&15)
  const int rb0  = tileM + ((lane >> 4) << 3);
  const int ncol = tileN + mrow;
  if (mode == 0) {
#pragma unroll
    for (int g = 0; g < 2; ++g)
#pragma unroll
      for (int r = 0; r < 8; ++r) {
        float* p = Cf + (size_t)(rb0 + 16 * g + r) * ldc + (size_t)ncol * colScale + colOff;
#pragma unroll
        for (int j = 0; j < 8; ++j) p[j * 16 * colScale] = acc[8 * g + j][r];
      }
  } else if (mode == 1) {
#pragma unroll
    for (int g = 0; g < 2; ++g)
#pragma unroll
      for (int r = 0; r < 8; ++r) {
        _Float16* p = Ch + (size_t)(rb0 + 16 * g + r) * ldc + (size_t)ncol * colScale + colOff;
#pragma unroll
        for (int j = 0; j < 8; ++j) p[j * 16 * colScale] = (_Float16)acc[8 * g + j][r];
      }
  } else if (mode == 2) {
    // c = ncol + 16j; b = c&3, y = c>>2 ; dst[(b*4096 + row)*128 + y]
#pragma unroll
    for (int j = 0; j < 8; ++j) {
      const int c = ncol + j * 16;
      _Float16* p = Ch + ((size_t)(c & 3) * 4096 + rb0) * 128 + (c >> 2);
#pragma unroll
      for (int g = 0; g < 2; ++g)
#pragma unroll
        for (int r = 0; r < 8; ++r)
          p[(size_t)(16 * g + r) * 128] = (_Float16)acc[8 * g + j][r];
    }
  } else {
    // dst[((c&3)*128 + (c>>2))*4096 + row]
#pragma unroll
    for (int j = 0; j < 8; ++j) {
      const int c = ncol + j * 16;
      _Float16* p = Ch + ((size_t)(c & 3) * 128 + (c >> 2)) * 4096 + rb0;
#pragma unroll
      for (int g = 0; g < 2; ++g)
#pragma unroll
        for (int r = 0; r < 8; ++r)
          p[16 * g + r] = (_Float16)acc[8 * g + j][r];
    }
  }
}

// ---------------------------------------------------------------------------
// Row prep: layernorm(h) -> hc[:,0:512] (f16), zero stat/pad cols;
//           L1-normalize x -> xn (f16) and xn^T (f16)
// ---------------------------------------------------------------------------
__global__ __launch_bounds__(256) void prep_rows(
    const float* __restrict__ h, const float* __restrict__ x,
    const float* __restrict__ gamma, const float* __restrict__ beta,
    _Float16* __restrict__ hc, _Float16* __restrict__ xnh,
    _Float16* __restrict__ xnT)
{
  __shared__ float sm[256];
  const int row = blockIdx.x, tid = threadIdx.x;
  const float* hr = h + (size_t)row * INF;
  const float v0 = hr[tid], v1 = hr[tid + 256];

  sm[tid] = v0 + v1; __syncthreads();
  for (int s = 128; s; s >>= 1) { if (tid < s) sm[tid] += sm[tid + s]; __syncthreads(); }
  const float mu = sm[0] * (1.0f / INF); __syncthreads();

  sm[tid] = v0 * v0 + v1 * v1; __syncthreads();
  for (int s = 128; s; s >>= 1) { if (tid < s) sm[tid] += sm[tid + s]; __syncthreads(); }
  const float var = sm[0] * (1.0f / INF) - mu * mu; __syncthreads();
  const float inv = rsqrtf(var + 1e-5f);

  _Float16* hcr = hc + (size_t)row * HCPAD;
  hcr[tid]       = (_Float16)((v0 - mu) * inv * gamma[tid]       + beta[tid]);
  hcr[tid + 256] = (_Float16)((v1 - mu) * inv * gamma[tid + 256] + beta[tid + 256]);
  if (tid < HCPAD - INF) hcr[INF + tid] = (_Float16)0.0f;

  const float xv = x[(size_t)row * DXC + tid];
  sm[tid] = fabsf(xv); __syncthreads();
  for (int s = 128; s; s >>= 1) { if (tid < s) sm[tid] += sm[tid + s]; __syncthreads(); }
  const float il1 = 1.0f / fmaxf(sm[0], 1e-12f);
  const float xn  = xv * il1;
  xnh[(size_t)row * DXC + tid]   = (_Float16)xn;
  xnT[(size_t)tid * NROWS + row] = (_Float16)xn;
}

// ---------------------------------------------------------------------------
// Transposed f16 weight convert with K zero-padding: dst[n*Kpad+k] = src[k*N+n]
// ---------------------------------------------------------------------------
__global__ __launch_bounds__(256) void cvt_t(
    const float* __restrict__ src, _Float16* __restrict__ dst,
    int Ksrc, int Nsrc, int Kpad)
{
  const int idx = blockIdx.x * 256 + threadIdx.x;
  if (idx >= Nsrc * Kpad) return;
  const int n = idx / Kpad, k = idx - n * Kpad;
  dst[idx] = (k < Ksrc) ? (_Float16)src[(size_t)k * Nsrc + n] : (_Float16)0.0f;
}

// softmax(mixing, axis=0): column-wise over 5x5
__global__ void mix_softmax_k(const float* __restrict__ mixing, float* __restrict__ msoft)
{
  const int c = threadIdx.x;
  if (c < 5) {
    float mx = -3.4e38f;
    for (int i = 0; i < 5; ++i) mx = fmaxf(mx, mixing[i * 5 + c]);
    float e[5], d = 0.0f;
    for (int i = 0; i < 5; ++i) { e[i] = __expf(mixing[i * 5 + c] - mx); d += e[i]; }
    for (int i = 0; i < 5; ++i) msoft[i * 5 + c] = e[i] / d;
  }
}

// ---------------------------------------------------------------------------
// a_x row stats (diag, sum, std ddof=1) -> hc cols 512..514; a_x -> chan0 f16
// ---------------------------------------------------------------------------
__global__ __launch_bounds__(256) void stats_rows(
    const float* __restrict__ S, _Float16* __restrict__ chan0,
    _Float16* __restrict__ hc)
{
  __shared__ float sm[256];
  const int row = blockIdx.x, tid = threadIdx.x;
  const float* r = S + (size_t)row * NROWS;
  _Float16* out = chan0 + (size_t)row * NROWS;
  float s = 0.0f, s2 = 0.0f;
  for (int i = tid; i < NROWS; i += 256) {
    const float v = r[i];
    s += v; s2 += v * v;
    out[i] = (_Float16)v;
  }
  sm[tid] = s; __syncthreads();
  for (int st = 128; st; st >>= 1) { if (tid < st) sm[tid] += sm[tid + st]; __syncthreads(); }
  const float sum = sm[0]; __syncthreads();
  sm[tid] = s2; __syncthreads();
  for (int st = 128; st; st >>= 1) { if (tid < st) sm[tid] += sm[tid + st]; __syncthreads(); }
  if (tid == 0) {
    const float mean = sum * (1.0f / NROWS);
    const float varp = (sm[0] - (float)NROWS * mean * mean) * (1.0f / (NROWS - 1));
    _Float16* hcr = hc + (size_t)row * HCPAD;
    hcr[512] = (_Float16)r[row];                   // diag
    hcr[513] = (_Float16)sum;                      // row sum
    hcr[514] = (_Float16)sqrtf(fmaxf(varp, 0.0f)); // sample std
  }
}

// Row softmax of raw scores (f32) -> f16 channel
__global__ __launch_bounds__(256) void softmax_rows(
    const float* __restrict__ S, _Float16* __restrict__ chan)
{
  __shared__ float sm[256];
  const int row = blockIdx.x, tid = threadIdx.x;
  const float* r = S + (size_t)row * NROWS;
  _Float16* out = chan + (size_t)row * NROWS;
  float m = -3.4e38f;
  for (int i = tid; i < NROWS; i += 256) m = fmaxf(m, r[i]);
  sm[tid] = m; __syncthreads();
  for (int s = 128; s; s >>= 1) { if (tid < s) sm[tid] = fmaxf(sm[tid], sm[tid + s]); __syncthreads(); }
  m = sm[0]; __syncthreads();
  float sum = 0.0f;
  for (int i = tid; i < NROWS; i += 256) {
    const float e = __expf(r[i] - m);
    sum += e;
    out[i] = (_Float16)e;
  }
  sm[tid] = sum; __syncthreads();
  for (int s = 128; s; s >>= 1) { if (tid < s) sm[tid] += sm[tid + s]; __syncthreads(); }
  const float inv = 1.0f / sm[0];
  for (int i = tid; i < NROWS; i += 256) out[i] = (_Float16)((float)out[i] * inv);
}

// A_col = sum_i chan_i * msoft[i][col]  (elementwise, 8 elems/thread)
__global__ __launch_bounds__(256) void combine_kernel(
    const _Float16* __restrict__ c0, const _Float16* __restrict__ c1,
    const _Float16* __restrict__ c2, const _Float16* __restrict__ c3,
    const _Float16* __restrict__ c4, const float* __restrict__ msoft,
    int col, _Float16* __restrict__ out)
{
  const float w0 = msoft[0 * 5 + col], w1 = msoft[1 * 5 + col],
              w2 = msoft[2 * 5 + col], w3 = msoft[3 * 5 + col],
              w4 = msoft[4 * 5 + col];
  const size_t base = ((size_t)blockIdx.x * 256 + threadIdx.x) * 8;
  v8h a0 = *(const v8h*)(c0 + base), a1 = *(const v8h*)(c1 + base),
      a2 = *(const v8h*)(c2 + base), a3 = *(const v8h*)(c3 + base),
      a4 = *(const v8h*)(c4 + base);
  v8h o;
#pragma unroll
  for (int i = 0; i < 8; ++i)
    o[i] = (_Float16)(w0 * (float)a0[i] + w1 * (float)a1[i] + w2 * (float)a2[i] +
                      w3 * (float)a3[i] + w4 * (float)a4[i]);
  *(v8h*)(out + base) = o;
}

// h_out = elu(F + bfc) + h0
__global__ __launch_bounds__(256) void final_h(
    const float* __restrict__ F, const float* __restrict__ bfc,
    const float* __restrict__ h0, float* __restrict__ out)
{
  const size_t idx = (size_t)blockIdx.x * 256 + threadIdx.x;
  const int c = (int)(idx & (HIDC - 1));
  float v = F[idx] + bfc[c];
  v = v > 0.0f ? v : (__expf(v) - 1.0f);
  out[idx] = v + h0[idx];
}

__global__ __launch_bounds__(256) void add_x0(
    float* __restrict__ xout, const float* __restrict__ x0)
{
  const size_t idx = (size_t)blockIdx.x * 256 + threadIdx.x;
  xout[idx] += x0[idx];
}

// ---------------------------------------------------------------------------
// Host orchestration
// ---------------------------------------------------------------------------
extern "C" void kernel_launch(void* const* d_in, const int* in_sizes, int n_in,
                              void* d_out, int out_size, void* d_ws, size_t ws_size,
                              hipStream_t stream) {
  (void)in_sizes; (void)n_in; (void)out_size; (void)ws_size;
  const float* h      = (const float*)d_in[0];
  const float* x      = (const float*)d_in[1];
  const float* gamma  = (const float*)d_in[2];
  const float* beta   = (const float*)d_in[3];
  const float* Wk     = (const float*)d_in[4];
  const float* Wq     = (const float*)d_in[5];
  const float* Wv     = (const float*)d_in[6];
  const float* Wfc    = (const float*)d_in[7];
  const float* bfc    = (const float*)d_in[8];
  const float* mixing = (const float*)d_in[9];
  float* out = (float*)d_out;

  // --- workspace carving (256B aligned) ---
  char* ws = (char*)d_ws;
  size_t off = 0;
  auto take = [&](size_t bytes) -> char* {
    off = (off + 255) & ~(size_t)255;
    char* p = ws + off;
    off += bytes;
    return p;
  };
  const size_t NN = (size_t)NROWS * NROWS;
  float*     Sraw  = (float*)take(NN * 4);              // raw scores / reused as F
  _Float16*  chan[5];
  for (int i = 0; i < 5; ++i) chan[i] = (_Float16*)take(NN * 2);
  _Float16*  Acomb = (_Float16*)take(NN * 2);
  _Float16*  hc    = (_Float16*)take((size_t)NROWS * HCPAD * 2);
  _Float16*  xnh   = (_Float16*)take((size_t)NROWS * DXC * 2);
  _Float16*  xnT   = (_Float16*)take((size_t)DXC * NROWS * 2);
  _Float16*  Wkt   = (_Float16*)take((size_t)HIDC * INF * 2);
  _Float16*  Wqt   = (_Float16*)take((size_t)HIDC * INF * 2);
  _Float16*  Wvt   = (_Float16*)take((size_t)HIDC * HCPAD * 2);
  _Float16*  Wfct  = (_Float16*)take((size_t)HIDC * HIDC * 2);
  _Float16*  Kh    = (_Float16*)take((size_t)4 * NROWS * 128 * 2);
  _Float16*  Qh    = (_Float16*)take((size_t)4 * NROWS * 128 * 2);
  _Float16*  Vt    = (_Float16*)take((size_t)4 * 128 * NROWS * 2);
  _Float16*  houth = (_Float16*)take((size_t)NROWS * HIDC * 2);
  float*     msoft = (float*)take(25 * 4);

  auto gemm = [&](const _Float16* A, int lda, const _Float16* Bt, int ldb,
                  int M, int Nn, int Kd, int mode,
                  float* Cf, _Float16* Ch, int ldc, int cs, int co) {
    dim3 g(M / 128, Nn / 128);
    wmma_gemm<<<g, dim3(128), 0, stream>>>(A, lda, Bt, ldb, Kd, mode, Cf, Ch, ldc, cs, co);
  };

  // 1) prep: layernorm + L1 norm
  prep_rows<<<NROWS, 256, 0, stream>>>(h, x, gamma, beta, hc, xnh, xnT);

  // 2) weight transposes -> f16
  cvt_t<<<(HIDC * INF + 255) / 256, 256, 0, stream>>>(Wk, Wkt, INF, HIDC, INF);
  cvt_t<<<(HIDC * INF + 255) / 256, 256, 0, stream>>>(Wq, Wqt, INF, HIDC, INF);
  cvt_t<<<(HIDC * HCPAD + 255) / 256, 256, 0, stream>>>(Wv, Wvt, INF + 3, HIDC, HCPAD);
  cvt_t<<<(HIDC * HIDC + 255) / 256, 256, 0, stream>>>(Wfc, Wfct, HIDC, HIDC, HIDC);

  // 3) mixing softmax (axis=0)
  mix_softmax_k<<<1, 32, 0, stream>>>(mixing, msoft);

  // 4) K, Q projections: hn @ Wk/Wq -> head-major f16
  gemm(hc, HCPAD, Wkt, INF, NROWS, HIDC, INF, 2, nullptr, Kh, 0, 0, 0);
  gemm(hc, HCPAD, Wqt, INF, NROWS, HIDC, INF, 2, nullptr, Qh, 0, 0, 0);

  // 5) a_x = xn @ xn^T (raw f32)
  gemm(xnh, DXC, xnh, DXC, NROWS, NROWS, DXC, 0, Sraw, nullptr, NROWS, 1, 0);

  // 6) row stats -> hc[:,512:515]; a_x -> chan0 (f16)
  stats_rows<<<NROWS, 256, 0, stream>>>(Sraw, chan[0], hc);

  // 7) V projection over padded K=576 -> per-head transposed f16
  gemm(hc, HCPAD, Wvt, HCPAD, NROWS, HIDC, HCPAD, 3, nullptr, Vt, 0, 0, 0);

  // 8) per-head scores + softmax -> chan[1+b]
  for (int b = 0; b < 4; ++b) {
    const _Float16* Kb = Kh + (size_t)b * NROWS * 128;
    const _Float16* Qb = Qh + (size_t)b * NROWS * 128;
    gemm(Kb, 128, Qb, 128, NROWS, NROWS, 128, 0, Sraw, nullptr, NROWS, 1, 0);
    softmax_rows<<<NROWS, 256, 0, stream>>>(Sraw, chan[1 + b]);
  }

  // 9) per-head mixed attention: A_{b+1} @ V_b -> hout (cols z*4+b)
  const int cgrid = (int)(NN / 2048);
  for (int b = 0; b < 4; ++b) {
    combine_kernel<<<cgrid, 256, 0, stream>>>(chan[0], chan[1], chan[2], chan[3],
                                              chan[4], msoft, b + 1, Acomb);
    const _Float16* Vb = Vt + (size_t)b * 128 * NROWS;
    gemm(Acomb, NROWS, Vb, NROWS, NROWS, 128, NROWS, 1, nullptr, houth, HIDC, 4, b);
  }

  // 10) x_out = A_0 @ xn (+ x0)
  combine_kernel<<<cgrid, 256, 0, stream>>>(chan[0], chan[1], chan[2], chan[3],
                                            chan[4], msoft, 0, Acomb);
  float* xout = out + (size_t)NROWS * HIDC;
  gemm(Acomb, NROWS, xnT, NROWS, NROWS, DXC, NROWS, 0, xout, nullptr, DXC, 1, 0);
  add_x0<<<(NROWS * DXC) / 256, 256, 0, stream>>>(xout, x);

  // 11) final: elu(hout @ Wfc + bfc) + h0  (reuse Sraw as f32 F buffer)
  gemm(houth, HIDC, Wfct, HIDC, NROWS, HIDC, HIDC, 0, Sraw, nullptr, HIDC, 1, 0);
  final_h<<<(NROWS * HIDC) / 256, 256, 0, stream>>>(Sraw, bfc, h, out);
}
